// SemanticEncoder_68590627717287
// MI455X (gfx1250) — matrix operand order repaired and checked by reading, not compiled, not run on hardware
//
#include <hip/hip_runtime.h>
#include <stdint.h>

// ---------------------------------------------------------------------------
// Problem constants (shape fixed by the reference: [3, 256, 96, 96])
// ---------------------------------------------------------------------------
#define NB   3
#define CCH  256
#define HW   9216          // 96*96
#define EPSV 1e-8f

// Phase-2 tiling
#define BM   128           // rows (query positions) per block = 8 waves * 16
#define BN   64            // key positions per LDS tile
#define BPAD 264           // padded LDS row stride (bf16) = 528B -> conflict-free b128 reads
#define TILE (BN * BPAD)   // bf16 elems per buffer

typedef __attribute__((ext_vector_type(16))) __bf16 v16bf;
typedef __attribute__((ext_vector_type(8)))  float  v8f;
typedef __attribute__((ext_vector_type(4)))  float  f32x4;

union BF16x16 { v16bf v; f32x4 f[2]; };

// ---------------------------------------------------------------------------
// Phase 1: per-position L2 normalization over channels + transpose to
// [n][pos][c] bf16 so that WMMA fragments become contiguous 16B chunks.
// ---------------------------------------------------------------------------
__global__ __launch_bounds__(256) void nt_kernel(const float* __restrict__ src,
                                                 __bf16* __restrict__ dst) {
    __shared__ float tile[CCH][33];   // padded
    __shared__ float psum[8][32];
    __shared__ float sinv[32];

    const int blocksPerBatch = HW / 32;               // 288
    const int n  = blockIdx.x / blocksPerBatch;
    const int p0 = (blockIdx.x % blocksPerBatch) * 32;
    const int t    = threadIdx.x;
    const int r    = t >> 5;                          // 0..7
    const int lane = t & 31;                          // position within tile

    const float* base = src + (size_t)n * CCH * HW + p0 + lane;
    float ss = 0.f;
    #pragma unroll 4
    for (int c = r; c < CCH; c += 8) {
        float v = base[(size_t)c * HW];               // coalesced across lanes
        tile[c][lane] = v;
        ss += v * v;
    }
    psum[r][lane] = ss;
    __syncthreads();
    if (r == 0) {
        float tot = 0.f;
        #pragma unroll
        for (int i = 0; i < 8; ++i) tot += psum[i][lane];
        float nrm = sqrtf(tot + EPSV);
        sinv[lane] = 1.0f / (nrm + EPSV);
    }
    __syncthreads();

    // write [pos][c] bf16, 8 threads cover one position (coalesced 512B rows)
    const int p  = t >> 3;                            // 0..31
    const int c0 = (t & 7) * 32;
    const float inv = sinv[p];
    __bf16* drow = dst + ((size_t)n * HW + p0 + p) * CCH + c0;
    #pragma unroll
    for (int j = 0; j < 32; ++j)
        drow[j] = (__bf16)(tile[c0 + j][p] * inv);
}

// ---------------------------------------------------------------------------
// Phase 2: fused  sim = Anᵀ·Bn  (bf16 WMMA, f32 accum)  +  running argmax.
// A strip resident in VGPRs; B tiles streamed with ASYNC global->LDS copies
// into a double-buffered LDS tile; one s_wait_asynccnt + barrier per iter.
// ---------------------------------------------------------------------------
__global__ __launch_bounds__(256) void simargmax_kernel(const __bf16* __restrict__ AnT,
                                                        const __bf16* __restrict__ BnT,
                                                        int* __restrict__ zbest) {
    __shared__ __bf16 bsh[2 * TILE];                  // 2 x 33,792 B

    const int tilesM = HW / BM;                       // 72
    const int n   = blockIdx.x / tilesM;
    const int m0  = (blockIdx.x % tilesM) * BM;
    const int wave = threadIdx.x >> 5;
    const int lane = threadIdx.x & 31;
    const int g    = lane >> 4;                       // half-wave group
    const int nn   = lane & 15;

    const __bf16* bbase = BnT + (size_t)n * HW * CCH;
    const unsigned lds_base = (unsigned)(uintptr_t)(&bsh[0]);   // LDS offset (flat low 32b)

    // per-thread staging geometry: 8 x 16B chunks per tile
    const int brow0 = threadIdx.x >> 5;               // +8 per step
    const int coff  = (threadIdx.x & 31) * 8;

    // async-copy one whole B tile [BN][CCH] into buffer `buf` (no VGPR data)
    auto stage_async = [&](int buf, int it) {
        const __bf16* gsrc = bbase + (size_t)it * BN * CCH + coff;
        unsigned ldst = lds_base + (unsigned)buf * (TILE * 2u) +
                        (unsigned)(brow0 * BPAD + coff) * 2u;
        #pragma unroll
        for (int i = 0; i < 8; ++i) {
            const __bf16* g = gsrc + (brow0 + i * 8) * CCH;
            unsigned l = ldst + (unsigned)(i * 8 * BPAD) * 2u;
            asm volatile("global_load_async_to_lds_b128 %0, %1, off"
                         :: "v"(l), "v"(g) : "memory");
        }
    };

    // prime the pipeline: tile 0 -> buffer 0 (overlapped with A-strip loads)
    stage_async(0, 0);

    // --- resident A strip: 16 rows x 256 ch per wave, in registers ---------
    // A-matrix 16x32 bf16 layout: lane holds M=lane&15; K chunks
    // {g*8..g*8+7} and {16+g*8..16+g*8+7} -> two contiguous 16B chunks.
    const int arow = m0 + wave * 16 + nn;
    const __bf16* ap = AnT + ((size_t)n * HW + arow) * CCH;
    f32x4 areg[8][2];
    #pragma unroll
    for (int kk = 0; kk < 8; ++kk) {
        areg[kk][0] = *(const f32x4*)(ap + kk * 32 + g * 8);
        areg[kk][1] = *(const f32x4*)(ap + kk * 32 + 16 + g * 8);
    }

    float bestv[8];
    int   besti[8];
    #pragma unroll
    for (int j = 0; j < 8; ++j) { bestv[j] = -3.0e38f; besti[j] = 0; }

    asm volatile("s_wait_asynccnt 0x0" ::: "memory");
    __syncthreads();

    const int ITERS = HW / BN;                        // 144
    for (int it = 0; it < ITERS; ++it) {
        const int buf = it & 1;

        // kick off next tile into the other buffer; overlaps with WMMAs below
        if (it + 1 < ITERS) stage_async(buf ^ 1, it + 1);

        const __bf16* bt = &bsh[buf * TILE];
        const int n0 = it * BN;

        v8f acc[4];
        #pragma unroll
        for (int tt = 0; tt < 4; ++tt) acc[tt] = (v8f){0,0,0,0,0,0,0,0};

        #pragma unroll
        for (int kk = 0; kk < 8; ++kk) {
            BF16x16 af;
            af.f[0] = areg[kk][0];
            af.f[1] = areg[kk][1];
            #pragma unroll
            for (int tt = 0; tt < 4; ++tt) {
                // B-matrix 32x16 layout: lane holds N=lane&15; K chunks
                // {g*8..g*8+7} and {16+g*8..16+g*8+7} -> two 16B chunks.
                BF16x16 bfr;
                const f32x4* bp =
                    (const f32x4*)(bt + (tt * 16 + nn) * BPAD + kk * 32 + g * 8);
                bfr.f[0] = bp[0];
                bfr.f[1] = bp[2];
                acc[tt] = __builtin_amdgcn_wmma_f32_16x16x32_bf16(
                    false, af.v, false, bfr.v, (short)0, acc[tt], false, false);
            }
        }

        // fused running-argmax epilogue: per-lane only, co-executes with XDL.
        // C/D layout: VGPR j holds (M = j + 8*g, N = nn).
        #pragma unroll
        for (int j = 0; j < 8; ++j) {
            float v = acc[0][j];
            int   c = n0 + nn;
            #pragma unroll
            for (int tt = 1; tt < 4; ++tt) {
                float w = acc[tt][j];
                if (w > v) { v = w; c = n0 + tt * 16 + nn; }   // strict: keep first idx
            }
            if (v > bestv[j]) { bestv[j] = v; besti[j] = c; }
        }

        // my async copies for it+1 done; all waves done reading this buffer
        asm volatile("s_wait_asynccnt 0x0" ::: "memory");
        __syncthreads();
    }

    // final cross-lane argmax within each 16-lane half (one row per (g, j))
    #pragma unroll
    for (int j = 0; j < 8; ++j) {
        float v = bestv[j];
        int   c = besti[j];
        #pragma unroll
        for (int off = 8; off > 0; off >>= 1) {
            float ov = __shfl_xor(v, off, 32);
            int   oc = __shfl_xor(c, off, 32);
            if (ov > v || (ov == v && oc < c)) { v = ov; c = oc; }
        }
        if (nn == 0) {
            int row = m0 + wave * 16 + j + 8 * g;
            zbest[n * HW + row] = c;
        }
    }
}

// ---------------------------------------------------------------------------
// Phase 3: gather matched style columns. style fits in 192MB L2.
// ---------------------------------------------------------------------------
__global__ __launch_bounds__(256) void gather_kernel(const float* __restrict__ s,
                                                     const int* __restrict__ zbest,
                                                     float* __restrict__ out) {
    size_t i = (size_t)blockIdx.x * 256 + threadIdx.x;
    int    k  = (int)(i % HW);
    size_t nc = i / HW;                 // n*C + c
    int    n  = (int)(nc / CCH);
    int idx = zbest[n * HW + k];
    out[i] = s[nc * HW + idx];
}

// ---------------------------------------------------------------------------
extern "C" void kernel_launch(void* const* d_in, const int* in_sizes, int n_in,
                              void* d_out, int out_size, void* d_ws, size_t ws_size,
                              hipStream_t stream) {
    const float* content       = (const float*)d_in[0];
    const float* content_style = (const float*)d_in[1];
    const float* style         = (const float*)d_in[2];

    const size_t elems = (size_t)NB * HW * CCH;       // 7,077,888
    __bf16* AnT = (__bf16*)d_ws;
    __bf16* BnT = AnT + elems;
    int*  zbest = (int*)(BnT + elems);

    nt_kernel<<<NB * (HW / 32), 256, 0, stream>>>(content, AnT);
    nt_kernel<<<NB * (HW / 32), 256, 0, stream>>>(content_style, BnT);
    simargmax_kernel<<<NB * (HW / BM), 256, 0, stream>>>(AnT, BnT, zbest);
    gather_kernel<<<(unsigned)((size_t)NB * CCH * HW / 256), 256, 0, stream>>>(
        style, zbest, (float*)d_out);
}
